// OTXCorr_39127152067010
// MI455X (gfx1250) — matrix operand order repaired and checked by reading, not compiled, not run on hardware
//
#include <hip/hip_runtime.h>
#include <hip/hip_bf16.h>
#include <hip/hip_fp16.h>

typedef _Float16 v16h __attribute__((ext_vector_type(16)));
typedef _Float16 v8h  __attribute__((ext_vector_type(8)));
typedef float    v8f  __attribute__((ext_vector_type(8)));

#define NBATCH 8
#define NCH    256
#define NP1    2048
#define NP2    1024
#define KNN_K  32
#define CIN    269
#define CPAD   288
#define EPSBN  1e-5f

// ---------------- CDNA5 async global->LDS helpers (inline asm; ISA ch.10/15.18) ----
// LDS byte address (u32) in one VGPR, 64-bit global address in a VGPR pair.
// Generic pointers to __shared__ truncate to the LDS offset (ISA 10.2 aperture rule).
__device__ __forceinline__ void async_ld_b128(const void* gptr, void* lptr)
{
  unsigned lds = (unsigned)(size_t)lptr;
  unsigned long long ga = (unsigned long long)(size_t)gptr;
  asm volatile("global_load_async_to_lds_b128 %0, %1, off"
               :: "v"(lds), "v"(ga) : "memory");
}
__device__ __forceinline__ void async_wait0()
{
  asm volatile("s_wait_asynccnt 0x0" ::: "memory");
}

// ---------------- workspace layout (bytes) ----------------
constexpr size_t SZ_SNT  = (size_t)NBATCH*NP1*NCH*2;     // f16 (N1,C) transposed
constexpr size_t SZ_TNT  = (size_t)NBATCH*NP2*NCH*2;
constexpr size_t SZ_K    = (size_t)NBATCH*NP1*NP2*4;     // 67 MB, fits 192MB L2
constexpr size_t SZ_U    = (size_t)NBATCH*NP1*4;
constexpr size_t SZ_R    = (size_t)NBATCH*NP1*4;
constexpr size_t SZ_CV   = (size_t)NBATCH*NP2*4;
constexpr size_t SZ_TAVG = (size_t)NBATCH*NCH*4;
constexpr size_t SZ_KV   = (size_t)NBATCH*NP1*KNN_K*4;
constexpr size_t SZ_KI   = (size_t)NBATCH*NP1*KNN_K*4;
constexpr size_t SZ_W1H  = (size_t)128*CPAD*2;
constexpr size_t SZ_W2H  = (size_t)256*128*2;
constexpr size_t SZ_B1   = 128*4;
constexpr size_t SZ_B2   = 256*4;

constexpr size_t OFF_SNT  = 0;
constexpr size_t OFF_TNT  = OFF_SNT  + SZ_SNT;
constexpr size_t OFF_K    = OFF_TNT  + SZ_TNT;
constexpr size_t OFF_U    = OFF_K    + SZ_K;
constexpr size_t OFF_R    = OFF_U    + SZ_U;
constexpr size_t OFF_CV   = OFF_R    + SZ_R;
constexpr size_t OFF_TAVG = OFF_CV   + SZ_CV;
constexpr size_t OFF_KV   = OFF_TAVG + SZ_TAVG;
constexpr size_t OFF_KI   = OFF_KV   + SZ_KV;
constexpr size_t OFF_W1H  = OFF_KI   + SZ_KI;
constexpr size_t OFF_W2H  = OFF_W1H  + SZ_W1H;
constexpr size_t OFF_B1   = OFF_W2H  + SZ_W2H;
constexpr size_t OFF_B2   = OFF_B1   + SZ_B1;

// ---------------- prep: fold BN into weights, init c ----------------
__global__ void k_prep(const float* __restrict__ W1, const float* __restrict__ g1,
                       const float* __restrict__ be1, const float* __restrict__ m1,
                       const float* __restrict__ v1,
                       const float* __restrict__ W2, const float* __restrict__ g2,
                       const float* __restrict__ be2, const float* __restrict__ m2,
                       const float* __restrict__ v2,
                       _Float16* __restrict__ W1h, _Float16* __restrict__ W2h,
                       float* __restrict__ b1f, float* __restrict__ b2f,
                       float* __restrict__ cvec)
{
  int tid = threadIdx.x;
  for (int i = tid; i < 128; i += 256) {
    float s = g1[i] * rsqrtf(v1[i] + EPSBN);
    b1f[i] = be1[i] - m1[i] * s;
  }
  for (int i = tid; i < 256; i += 256) {
    float s = g2[i] * rsqrtf(v2[i] + EPSBN);
    b2f[i] = be2[i] - m2[i] * s;
  }
  for (int i = tid; i < 128*CPAD; i += 256) {
    int row = i / CPAD, col = i % CPAD;
    float s = g1[row] * rsqrtf(v1[row] + EPSBN);
    float val = (col < CIN) ? W1[row*CIN + col] * s : 0.0f;
    W1h[i] = (_Float16)val;
  }
  for (int i = tid; i < 256*128; i += 256) {
    int row = i >> 7;
    float s = g2[row] * rsqrtf(v2[row] + EPSBN);
    W2h[i] = (_Float16)(W2[i] * s);
  }
  for (int i = tid; i < NBATCH*NP2; i += 256) cvec[i] = 1.0f;
}

// ---------------- normalize fmap1 -> snT f16 (b,n,c) ----------------
__global__ void k_norm1(const float* __restrict__ fmap1, _Float16* __restrict__ snT)
{
  int idx = blockIdx.x*256 + threadIdx.x;          // b*NP1+n
  int b = idx >> 11, n = idx & (NP1-1);
  const float* f = fmap1 + (size_t)b*NCH*NP1 + n;
  float ss = 0.0f;
  for (int c = 0; c < NCH; ++c) { float x = f[(size_t)c*NP1]; ss += x*x; }
  float inv = rsqrtf(ss);
  _Float16* o = snT + (size_t)idx*NCH;
  for (int c = 0; c < NCH; ++c) o[c] = (_Float16)(f[(size_t)c*NP1] * inv);
}

__global__ void k_norm2(const float* __restrict__ fmap2, _Float16* __restrict__ tnT)
{
  int idx = blockIdx.x*256 + threadIdx.x;          // b*NP2+m
  int b = idx >> 10, n = idx & (NP2-1);
  const float* f = fmap2 + (size_t)b*NCH*NP2 + n;
  float ss = 0.0f;
  for (int c = 0; c < NCH; ++c) { float x = f[(size_t)c*NP2]; ss += x*x; }
  float inv = rsqrtf(ss);
  _Float16* o = tnT + (size_t)idx*NCH;
  for (int c = 0; c < NCH; ++c) o[c] = (_Float16)(f[(size_t)c*NP2] * inv);
}

// ---------------- template average ----------------
__global__ void k_tavg(const float* __restrict__ fmap2, float* __restrict__ tavg)
{
  int b = blockIdx.x, c = threadIdx.x;
  const float* f = fmap2 + ((size_t)b*NCH + c)*NP2;
  float s = 0.0f;
  for (int m = 0; m < NP2; ++m) s += f[m];
  tavg[b*NCH + c] = s * (1.0f/NP2);
}

// ---------------- attention -> u ----------------
__global__ void k_u(const float* __restrict__ fmap1, const float* __restrict__ tavg,
                    float* __restrict__ u)
{
  __shared__ float ta[NCH];
  __shared__ float red[256];
  int b = blockIdx.x, tid = threadIdx.x;
  ta[tid] = tavg[b*NCH + tid];
  __syncthreads();
  const float* f = fmap1 + (size_t)b*NCH*NP1;
  float lsum = 0.0f;
  for (int n = tid; n < NP1; n += 256) {
    float acc = 0.0f;
    for (int c = 0; c < NCH; ++c) acc += ta[c] * f[(size_t)c*NP1 + n];
    acc = fmaxf(acc, 0.0f);
    u[b*NP1 + n] = acc;
    lsum += acc;
  }
  red[tid] = lsum; __syncthreads();
  for (int s = 128; s > 0; s >>= 1) {
    if (tid < s) red[tid] += red[tid + s];
    __syncthreads();
  }
  float inv = 1.0f / (red[0] + 1e-6f);
  for (int n = tid; n < NP1; n += 256) u[b*NP1 + n] *= inv;
}

// ---------------- similarity + cost + exp -> K (WMMA f16, A staged via async LDS) --
__global__ void k_sim(const _Float16* __restrict__ snT, const _Float16* __restrict__ tnT,
                      const float* __restrict__ xyz1, const float* __restrict__ xyz2,
                      float* __restrict__ Kmat)
{
  __shared__ __align__(16) _Float16 at[16][NCH];   // 8KB: 16 rows of snT, shared by 8 waves
  int tid = threadIdx.x;
  int lane = tid & 31, wave = tid >> 5;
  int bid = blockIdx.x;
  int b   = bid >> 10;
  int rem = bid & 1023;
  int n0  = (rem >> 3) << 4;                     // 128 n-tiles of 16
  int m0  = ((rem & 7) << 7) + (wave << 4);      // 8 groups of 128 cols, wave picks 16

  // async-stage the contiguous 8KB A-tile (16 rows x 256 f16) into LDS
  const char* abase = (const char*)(snT + (size_t)(b*NP1 + n0) * NCH);
  char* lbase = (char*)&at[0][0];
  for (int i = tid; i < 512; i += 256)
    async_ld_b128(abase + (size_t)i*16, lbase + (size_t)i*16);
  async_wait0();
  __syncthreads();

  const _Float16* arow = &at[lane & 15][0];
  const _Float16* brow = tnT + (size_t)(b*NP2 + m0 + (lane & 15)) * NCH;
  int koffA = (lane >> 4) << 3;                  // 0 / 8
  int koffB = (lane >> 4) << 4;                  // 0 / 16
  union { v8f v; float f[8]; } acc = {};
  union { v16h v; v8h h[2]; } a;
  for (int k0 = 0; k0 < NCH; k0 += 32) {
    a.h[0] = *(const v8h*)(arow + k0 + koffB);
    a.h[1] = *(const v8h*)(arow + k0 + 8 + koffB);
    v16h bf = *(const v16h*)(brow + k0 + koffB);
    acc.v = __builtin_amdgcn_wmma_f32_16x16x32_f16(false, a.v, false, bf,
                                                   (short)0, acc.v, false, false);
  }
  int m = m0 + (lane & 15);
  const float* p2 = xyz2 + ((size_t)b*NP2 + m)*3;
  float p2x = p2[0], p2y = p2[1], p2z = p2[2];
  int nbase = n0 + ((lane >> 4) << 3);
  for (int r = 0; r < 8; ++r) {
    int n = nbase + r;
    const float* p1 = xyz1 + ((size_t)b*NP1 + n)*3;
    float dx = p1[0]-p2x, dy = p1[1]-p2y, dz = p1[2]-p2z;
    float g = sqrtf(dx*dx + dy*dy + dz*dz);
    float cost = 1.0f - acc.f[r] + 0.1f*g;
    cost = fminf(fmaxf(cost, 0.0f), 1.0f);
    Kmat[((size_t)(b*NP1 + n) << 10) + m] = __expf(-10.0f * cost);
  }
}

// ---------------- sinkhorn: r update (wave per row; c async-staged to LDS) ---------
__global__ void k_sink_r(const float* __restrict__ Kmat, const float* __restrict__ cvec,
                         const float* __restrict__ u, float* __restrict__ rvec)
{
  __shared__ __align__(16) float cs[NP2];
  int tid = threadIdx.x, lane = tid & 31, wave = tid >> 5;
  int row0 = blockIdx.x << 3;
  int b = row0 >> 11;
  async_ld_b128(cvec + (size_t)b*NP2 + 4*tid, &cs[4*tid]);   // 1024 f32 = 256 x b128
  async_wait0();
  __syncthreads();
  int row = row0 + wave;
  const float* kr = Kmat + ((size_t)row << 10);
  float acc = 0.0f;
  for (int m = lane; m < NP2; m += 32) acc += kr[m] * cs[m];
  for (int off = 16; off; off >>= 1) acc += __shfl_xor(acc, off, 32);
  if (lane == 0) rvec[row] = u[row] / acc;
}

// ---------------- sinkhorn: c update (thread per column; r async-staged) -----------
__global__ void k_sink_c(const float* __restrict__ Kmat, const float* __restrict__ rvec,
                         float* __restrict__ cvec)
{
  __shared__ __align__(16) float rs[NP1];
  int tid = threadIdx.x;
  int b = blockIdx.x >> 2;
  int m = ((blockIdx.x & 3) << 8) + tid;
  async_ld_b128(rvec + (size_t)b*NP1 + 4*tid,       &rs[4*tid]);
  async_ld_b128(rvec + (size_t)b*NP1 + 4*tid + 1024, &rs[4*tid + 1024]);
  async_wait0();
  __syncthreads();
  const float* kb = Kmat + ((size_t)(b*NP1) << 10) + m;
  float acc = 0.0f;
  for (int n = 0; n < NP1; ++n) {
    if ((n & 31) == 0) __builtin_prefetch(kb + ((size_t)(n + 32) << 10), 0, 0);
    acc += kb[(size_t)n << 10] * rs[n];
  }
  cvec[b*NP2 + m] = (1.0f/NP2) / acc;
}

// ---------------- T = clip(r*c*K) + top-32 per row ----------------
__global__ void k_topk(const float* __restrict__ Kmat, const float* __restrict__ rvec,
                       const float* __restrict__ cvec,
                       float* __restrict__ knn_val, int* __restrict__ knn_idx)
{
  __shared__ float tv[NP2];
  __shared__ float sv[256];
  __shared__ int   si[256];
  int tid = threadIdx.x;
  int row = blockIdx.x;                 // b*NP1+n
  int b = row >> 11;
  float rv = rvec[row];
  const float* kr = Kmat + ((size_t)row << 10);
  for (int j = tid; j < NP2; j += 256) {
    float t = rv * cvec[b*NP2 + j] * kr[j];
    tv[j] = fminf(fmaxf(t, 1e-7f), 1.0f);
  }
  __syncthreads();
  for (int pick = 0; pick < KNN_K; ++pick) {
    float best = -1.0f; int bi = 0;
    for (int j = tid; j < NP2; j += 256)
      if (tv[j] > best) { best = tv[j]; bi = j; }
    sv[tid] = best; si[tid] = bi;
    __syncthreads();
    for (int s = 128; s > 0; s >>= 1) {
      if (tid < s && sv[tid + s] > sv[tid]) { sv[tid] = sv[tid + s]; si[tid] = si[tid + s]; }
      __syncthreads();
    }
    if (tid == 0) {
      knn_val[(size_t)row*KNN_K + pick] = sv[0];
      knn_idx[(size_t)row*KNN_K + pick] = si[0];
      tv[si[0]] = -2.0f;
    }
    __syncthreads();
  }
}

// ---------------- gather + 2-layer WMMA MLP + max-k + output GEMV ----------------
__global__ void k_mlp(const float* __restrict__ fmap2, const float* __restrict__ xyz2,
                      const float* __restrict__ bc2,
                      const float* __restrict__ knn_val, const int* __restrict__ knn_idx,
                      const _Float16* __restrict__ W1h, const _Float16* __restrict__ W2h,
                      const float* __restrict__ b1f, const float* __restrict__ b2f,
                      const float* __restrict__ Wout, const float* __restrict__ bout,
                      float* __restrict__ out)
{
  __shared__ __align__(32) _Float16 Xt[KNN_K][CPAD];   // gathered features, transposed
  __shared__ __align__(32) _Float16 h1t[KNN_K][128];   // layer-1 activations, transposed
  __shared__ float h2[256][KNN_K];
  __shared__ float hmax[256];
  __shared__ int   nbi[KNN_K];
  int tid = threadIdx.x, lane = tid & 31, wave = tid >> 5;
  int row = blockIdx.x;                 // b*NP1+n
  int b = row >> 11, n = row & (NP1-1);

  if (tid < KNN_K) {
    nbi[tid] = knn_idx[(size_t)row*KNN_K + tid];
    Xt[tid][0] = (_Float16)knn_val[(size_t)row*KNN_K + tid];
  }
  __syncthreads();
  for (int t = tid; t < 12*KNN_K; t += 256) {
    int ch = t >> 5, k = t & 31;
    int nb = nbi[k];
    float val = (ch < 3) ? xyz2[((size_t)b*NP2 + nb)*3 + ch]
                         : bc2[((size_t)b*NP2 + nb)*9 + (ch - 3)];
    Xt[k][1 + ch] = (_Float16)val;
  }
  for (int c0 = 0; c0 < NCH; c0 += 8) {
    int cl = tid >> 5, k = tid & 31;
    int c = c0 + cl;
    Xt[k][13 + c] = (_Float16)fmap2[((size_t)b*NCH + c)*NP2 + nbi[k]];
  }
  for (int t = tid; t < KNN_K*(CPAD-CIN); t += 256) {
    int k = t / (CPAD-CIN);
    int c = CIN + t % (CPAD-CIN);
    Xt[k][c] = (_Float16)0.0f;
  }
  __syncthreads();

  int koffA = (lane >> 4) << 3;
  int koffB = (lane >> 4) << 4;

  // layer 1: 128x32 = W1'(128x288) x X(288x32), 16 tiles / 8 waves
  for (int i = 0; i < 2; ++i) {
    int tt = wave + (i << 3);
    int mt = tt >> 1, nt = tt & 1;
    const _Float16* wr = W1h + (size_t)((mt << 4) + (lane & 15)) * CPAD;
    int col = (nt << 4) + (lane & 15);
    union { v8f v; float f[8]; } acc = {};
    union { v16h v; v8h h[2]; } a;
    for (int k0 = 0; k0 < CPAD; k0 += 32) {
      a.h[0] = *(const v8h*)(wr + k0 + koffB);
      a.h[1] = *(const v8h*)(wr + k0 + 8 + koffB);
      v16h bf = *(const v16h*)(&Xt[col][k0 + koffB]);
      acc.v = __builtin_amdgcn_wmma_f32_16x16x32_f16(false, a.v, false, bf,
                                                     (short)0, acc.v, false, false);
    }
    int mbase = (mt << 4) + ((lane >> 4) << 3);
    for (int r = 0; r < 8; ++r) {
      int m = mbase + r;
      h1t[col][m] = (_Float16)fmaxf(acc.f[r] + b1f[m], 0.0f);
    }
  }
  __syncthreads();

  // layer 2: 256x32 = W2'(256x128) x h1(128x32), 32 tiles / 8 waves
  for (int i = 0; i < 4; ++i) {
    int tt = wave + (i << 3);
    int mt = tt >> 1, nt = tt & 1;
    const _Float16* wr = W2h + (size_t)((mt << 4) + (lane & 15)) * 128;
    int col = (nt << 4) + (lane & 15);
    union { v8f v; float f[8]; } acc = {};
    union { v16h v; v8h h[2]; } a;
    for (int k0 = 0; k0 < 128; k0 += 32) {
      a.h[0] = *(const v8h*)(wr + k0 + koffB);
      a.h[1] = *(const v8h*)(wr + k0 + 8 + koffB);
      v16h bf = *(const v16h*)(&h1t[col][k0 + koffB]);
      acc.v = __builtin_amdgcn_wmma_f32_16x16x32_f16(false, a.v, false, bf,
                                                     (short)0, acc.v, false, false);
    }
    int mbase = (mt << 4) + ((lane >> 4) << 3);
    for (int r = 0; r < 8; ++r) {
      int m = mbase + r;
      h2[m][col] = fmaxf(acc.f[r] + b2f[m], 0.0f);
    }
  }
  __syncthreads();

  {
    float mx = h2[tid][0];
    for (int j = 1; j < KNN_K; ++j) mx = fmaxf(mx, h2[tid][j]);
    hmax[tid] = mx;
  }
  __syncthreads();

  if (tid < 32) {
    float acc = bout[tid];
    const float* w = Wout + tid*256;
    for (int c = 0; c < 256; ++c) acc += w[c] * hmax[c];
    out[((size_t)b*32 + tid)*NP1 + n] = acc;
  }
}

// ---------------- host launcher ----------------
extern "C" void kernel_launch(void* const* d_in, const int* in_sizes, int n_in,
                              void* d_out, int out_size, void* d_ws, size_t ws_size,
                              hipStream_t stream)
{
  const float* fmap1 = (const float*)d_in[0];
  const float* fmap2 = (const float*)d_in[1];
  const float* xyz1  = (const float*)d_in[2];
  const float* xyz2  = (const float*)d_in[3];
  const float* bc2   = (const float*)d_in[4];
  const float* W1    = (const float*)d_in[5];
  const float* g1    = (const float*)d_in[6];
  const float* be1   = (const float*)d_in[7];
  const float* m1    = (const float*)d_in[8];
  const float* v1    = (const float*)d_in[9];
  const float* W2    = (const float*)d_in[10];
  const float* g2    = (const float*)d_in[11];
  const float* be2   = (const float*)d_in[12];
  const float* m2    = (const float*)d_in[13];
  const float* v2    = (const float*)d_in[14];
  const float* Wout  = (const float*)d_in[15];
  const float* bout  = (const float*)d_in[16];

  char* ws = (char*)d_ws;
  _Float16* snT  = (_Float16*)(ws + OFF_SNT);
  _Float16* tnT  = (_Float16*)(ws + OFF_TNT);
  float*    Kmat = (float*)(ws + OFF_K);
  float*    u    = (float*)(ws + OFF_U);
  float*    rvec = (float*)(ws + OFF_R);
  float*    cvec = (float*)(ws + OFF_CV);
  float*    tavg = (float*)(ws + OFF_TAVG);
  float*    kval = (float*)(ws + OFF_KV);
  int*      kidx = (int*)(ws + OFF_KI);
  _Float16* W1h  = (_Float16*)(ws + OFF_W1H);
  _Float16* W2h  = (_Float16*)(ws + OFF_W2H);
  float*    b1f  = (float*)(ws + OFF_B1);
  float*    b2f  = (float*)(ws + OFF_B2);

  k_prep<<<1, 256, 0, stream>>>(W1, g1, be1, m1, v1, W2, g2, be2, m2, v2,
                                W1h, W2h, b1f, b2f, cvec);
  k_norm1<<<(NBATCH*NP1)/256, 256, 0, stream>>>(fmap1, snT);
  k_norm2<<<(NBATCH*NP2)/256, 256, 0, stream>>>(fmap2, tnT);
  k_tavg<<<NBATCH, 256, 0, stream>>>(fmap2, tavg);
  k_u<<<NBATCH, 256, 0, stream>>>(fmap1, tavg, u);
  k_sim<<<NBATCH*1024, 256, 0, stream>>>(snT, tnT, xyz1, xyz2, Kmat);

  for (int it = 0; it < 100; ++it) {
    k_sink_r<<<(NBATCH*NP1)/8, 256, 0, stream>>>(Kmat, cvec, u, rvec);
    k_sink_c<<<NBATCH*(NP2/256), 256, 0, stream>>>(Kmat, rvec, cvec);
  }

  k_topk<<<NBATCH*NP1, 256, 0, stream>>>(Kmat, rvec, cvec, kval, kidx);
  k_mlp<<<NBATCH*NP1, 256, 0, stream>>>(fmap2, xyz2, bc2, kval, kidx,
                                        W1h, W2h, b1f, b2f, Wout, bout,
                                        (float*)d_out);
}
